// OrthoLinear_81509889343648
// MI455X (gfx1250) — matrix operand order repaired and checked by reading, not compiled, not run on hardware
//
#include <hip/hip_runtime.h>
#include <hip/hip_fp16.h>

typedef __attribute__((ext_vector_type(16))) _Float16 v16h;
typedef __attribute__((ext_vector_type(8)))  _Float16 v8h;
typedef __attribute__((ext_vector_type(4)))  _Float16 v4h;
typedef __attribute__((ext_vector_type(8)))  float    v8f;
typedef __attribute__((ext_vector_type(4)))  float    v4f;

constexpr int KDIM = 4096;   // IN_FEATURES
constexpr int NDIM = 4096;   // OUT_FEATURES
constexpr int MDIM = 4096;   // N_TOKENS
constexpr int BM = 128, BN = 128, BK = 32;
constexpr int LDT = BK + 8;  // padded LDS row stride in halves (80B, 16B-aligned, bank-conflict-free)

// ---------------------------------------------------------------------------
// Dense path: Y = X(fp32->fp16) @ W^T using v_wmma_f32_16x16x32_f16.
// 256 threads = 8 waves; each wave computes a 32x64 sub-tile (2x4 WMMA tiles).
// ---------------------------------------------------------------------------
__global__ __launch_bounds__(256)
void gemm_wmma_f16(const float* __restrict__ X, const _Float16* __restrict__ W,
                   float* __restrict__ Y) {
  __shared__ __align__(16) _Float16 As[BM * LDT];
  __shared__ __align__(16) _Float16 Bs[BN * LDT];

  const int tid   = threadIdx.x;
  const int wave  = tid >> 5;
  const int lane  = tid & 31;
  const int lhalf = lane >> 4;   // 0 or 1 (half-wave)
  const int l16   = lane & 15;

  const int m0 = blockIdx.y * BM;
  const int n0 = blockIdx.x * BN;
  const int wm = wave & 3;       // 0..3 -> 32-row band
  const int wn = wave >> 2;      // 0..1 -> 64-col band

  v8f acc[2][4] = {};            // 8 x (16x16 f32) accumulators

  for (int k0 = 0; k0 < KDIM; k0 += BK) {
    // ---- stage X tile (128x32 fp32 -> fp16), 4 float4 segments per thread
#pragma unroll
    for (int i = 0; i < 4; ++i) {
      int seg = tid + i * 256;                 // 0..1023
      int row = seg >> 3;                      // 128 rows, 8 segs/row
      int kc  = (seg & 7) * 4;
      v4f f = *(const v4f*)(X + (size_t)(m0 + row) * KDIM + k0 + kc);
      v4h h = { (_Float16)f.x, (_Float16)f.y, (_Float16)f.z, (_Float16)f.w };
      *(v4h*)(&As[row * LDT + kc]) = h;        // ds_store_b64
    }
    // ---- stage W tile (128x32 fp16), 2 x 16B segments per thread
#pragma unroll
    for (int i = 0; i < 2; ++i) {
      int seg = tid + i * 256;                 // 0..511
      int row = seg >> 2;                      // 128 rows, 4 segs/row
      int kc  = (seg & 3) * 8;
      v8h h = *(const v8h*)(W + (size_t)(n0 + row) * KDIM + k0 + kc);
      *(v8h*)(&Bs[row * LDT + kc]) = h;        // ds_store_b128
    }
    // prefetch next K tile into cache hierarchy (global_prefetch_b8)
    if (k0 + BK < KDIM) {
      __builtin_prefetch(X + (size_t)(m0 + (tid >> 1)) * KDIM + k0 + BK, 0, 3);
      __builtin_prefetch(W + (size_t)(n0 + (tid >> 1)) * KDIM + k0 + BK, 0, 3);
    }
    __syncthreads();

    // ---- build fragments from LDS (layouts per ISA 7.12.2)
    v16h afrag[2], bfrag[4];
#pragma unroll
    for (int mt = 0; mt < 2; ++mt) {
      int row = wm * 32 + mt * 16 + l16;
      int kb  = lhalf * 8;                       // lanes 16-31 take K+8 block
      v8h lo = *(const v8h*)(&As[row * LDT + kb]);       // K = kb..kb+7
      v8h hi = *(const v8h*)(&As[row * LDT + 16 + kb]);  // K = 16+kb..
      v16h a;
      *(v8h*)&a       = lo;
      *((v8h*)&a + 1) = hi;
      afrag[mt] = a;
    }
#pragma unroll
    for (int nt = 0; nt < 4; ++nt) {
      int col = wn * 64 + nt * 16 + l16;
      int kb  = lhalf * 8;                       // lanes 16-31 take K+8 block
      v8h lo = *(const v8h*)(&Bs[col * LDT + kb]);
      v8h hi = *(const v8h*)(&Bs[col * LDT + 16 + kb]);
      v16h b;
      *(v8h*)&b       = lo;
      *((v8h*)&b + 1) = hi;
      bfrag[nt] = b;
    }

    // ---- 8 WMMAs per wave per K-step
#pragma unroll
    for (int mt = 0; mt < 2; ++mt)
#pragma unroll
      for (int nt = 0; nt < 4; ++nt)
        acc[mt][nt] = __builtin_amdgcn_wmma_f32_16x16x32_f16(
            false, afrag[mt], false, bfrag[nt],
            (short)0, acc[mt][nt], false, false);

    __syncthreads();
  }

  // ---- epilogue: C/D layout -> lanes 0-15 hold M=r, lanes 16-31 hold M=8+r
#pragma unroll
  for (int mt = 0; mt < 2; ++mt) {
    int mbase = m0 + wm * 32 + mt * 16 + lhalf * 8;
#pragma unroll
    for (int nt = 0; nt < 4; ++nt) {
      int col = n0 + wn * 64 + nt * 16 + l16;
#pragma unroll
      for (int r = 0; r < 8; ++r)
        Y[(size_t)(mbase + r) * NDIM + col] = acc[mt][nt][r];
    }
  }
}

// ---------------------------------------------------------------------------
// Sparse COO correction: out[b, r] += x[b, c] * v  (ALPHA == 1.0)
// one block per nnz; 256 threads x 4 tokens each; grid.y covers 4096 tokens
// ---------------------------------------------------------------------------
__global__ __launch_bounds__(256)
void ortho_scatter(const float* __restrict__ X, const float* __restrict__ vals,
                   const int* __restrict__ idx, float* __restrict__ Y) {
  int j = blockIdx.x;
  int e = idx[j];
  int r = e >> 12;          // / IN_FEATURES (4096)
  int c = e & (KDIM - 1);   // % IN_FEATURES
  float v = vals[j];        // * ALPHA (1.0)
  int b0 = blockIdx.y * 1024 + threadIdx.x;
#pragma unroll
  for (int t = 0; t < 4; ++t) {
    int b = b0 + t * 256;
    atomicAdd(Y + (size_t)b * NDIM + r, X[(size_t)b * KDIM + c] * v);
  }
}

extern "C" void kernel_launch(void* const* d_in, const int* in_sizes, int n_in,
                              void* d_out, int out_size, void* d_ws, size_t ws_size,
                              hipStream_t stream) {
  const float*    x    = (const float*)d_in[0];
  const _Float16* w    = (const _Float16*)d_in[1];   // base_weight is fp16
  const float*    vals = (const float*)d_in[2];
  const int*      idx  = (const int*)d_in[3];
  float*          out  = (float*)d_out;

  dim3 grid(NDIM / BN, MDIM / BM);                   // 32 x 32 tiles
  gemm_wmma_f16<<<grid, 256, 0, stream>>>(x, w, out);

  int nnz = in_sizes[2];                             // 32768
  dim3 grid2(nnz, MDIM / 1024);                      // (32768, 4)
  ortho_scatter<<<grid2, 256, 0, stream>>>(x, vals, idx, out);
}
